// MambaBlocks_27427661153048
// MI455X (gfx1250) — matrix operand order repaired and checked by reading, not compiled, not run on hardware
//
#include <hip/hip_runtime.h>
#include <hip/hip_bf16.h>

// ---------------------------------------------------------------------------
// CDNA5 WMMA types (wave32): 16x16x32 bf16 -> f32 accumulate
// ---------------------------------------------------------------------------
typedef __attribute__((ext_vector_type(16))) __bf16 v16bf;
typedef __attribute__((ext_vector_type(8)))  float  v8f;
typedef int v4i __attribute__((vector_size(16)));

union FragBF {
    v16bf    v;
    unsigned u[8];
};

#define AS1 __attribute__((address_space(1)))
#define AS3 __attribute__((address_space(3)))

#if defined(__has_builtin)
#if __has_builtin(__builtin_amdgcn_global_load_async_to_lds_b128)
#define USE_ASYNC_LDS 1
#endif
#endif

__device__ __forceinline__ unsigned short f2bf(float f) {
    unsigned u = __float_as_uint(f);
    // round-to-nearest-even f32 -> bf16
    return (unsigned short)((u + 0x7FFFu + ((u >> 16) & 1u)) >> 16);
}

__device__ __forceinline__ unsigned pack_bf16(float f0, float f1) {
    return (unsigned)f2bf(f0) | ((unsigned)f2bf(f1) << 16);
}

__device__ __forceinline__ float softplusf(float x) {
    return (x > 20.f) ? x : __logf(1.f + __expf(x));
}

// ---------------------------------------------------------------------------
// f32 -> bf16 bulk converter (for weights). n2 = number of element PAIRS.
// ---------------------------------------------------------------------------
__global__ __launch_bounds__(256)
void cvt_f32_bf16_kernel(const float* __restrict__ src,
                         unsigned short* __restrict__ dst, int n2)
{
    int i = blockIdx.x * 256 + threadIdx.x;
    if (i < n2) {
        *(unsigned*)&dst[2 * i] = pack_bf16(src[2 * i], src[2 * i + 1]);
    }
}

// ---------------------------------------------------------------------------
// GEMM: C[M,N] = A[M,K] * W[N,K]^T, A and W already bf16, f32 accumulate.
// mode==1: C = softplus(acc + bias[n]).  Optional bf16 side-store Cbf for the
// first Nbf columns (feeds the next GEMM without a convert pass).
// Tile 64x128 per 256-thread block; 8 waves in 4(M) x 2(N); each wave computes
// a 16x64 strip: ONE A fragment reused by FOUR WMMAs per K-step (K step 32).
// Staging: async global->LDS B128 chunks (ASYNCcnt path).
// Requires M%64==0, K%32==0; N ragged (weight row clamped, stores guarded).
// ---------------------------------------------------------------------------
#define GBM 64
#define GBN 128
#define GKT 32
#define GLT 40   // LDS row stride (bf16 elems): 80B rows, 16B aligned, conflict-free

__global__ __launch_bounds__(256)
void gemm_bf16_wmma(const unsigned short* __restrict__ A, int lda,
                    const unsigned short* __restrict__ W,    // N x K bf16
                    const float* __restrict__ bias,          // may be null
                    float* __restrict__ C, int ldc,
                    unsigned short* __restrict__ Cbf, int ldcbf, int Nbf,
                    int M, int N, int K, int mode)
{
    __shared__ unsigned short As[GBM * GLT];   //  5 KB
    __shared__ unsigned short Bs[GBN * GLT];   // 10 KB

    const int tid  = threadIdx.x;
    const int m0   = blockIdx.x * GBM;
    const int n0   = blockIdx.y * GBN;
    const int lane = tid & 31;
    const int w    = tid >> 5;
    const int wm   = w & 3;       // 0..3 : M sub-tile
    const int wn   = w >> 2;      // 0..1 : 64-wide N strip
    const int hi   = lane >> 4;
    const int l15  = lane & 15;

    v8f acc[4];
    #pragma unroll
    for (int g = 0; g < 4; ++g) acc[g] = (v8f){};

    // staging: thread t copies 16B (8 bf16): row = t/4, kbase = (t%4)*8
    // A tile: 64 rows (1 chunk/thread). B tile: 128 rows (2 chunks/thread).
    const int srow = tid >> 2;
    const int kb   = (tid & 3) * 8;
    int wrow0 = n0 + srow;      if (wrow0 >= N) wrow0 = N - 1;   // clamp ragged N
    int wrow1 = n0 + srow + 64; if (wrow1 >= N) wrow1 = N - 1;
    const unsigned short* ga  = A + (size_t)(m0 + srow) * lda + kb;
    const unsigned short* gb0 = W + (size_t)wrow0 * K + kb;
    const unsigned short* gb1 = W + (size_t)wrow1 * K + kb;
    unsigned short* la  = &As[srow * GLT + kb];
    unsigned short* lb0 = &Bs[srow * GLT + kb];
    unsigned short* lb1 = &Bs[(srow + 64) * GLT + kb];

    const int arow = (wm * 16 + l15) * GLT;
    int brow[4];
    #pragma unroll
    for (int g = 0; g < 4; ++g) brow[g] = (wn * 64 + g * 16 + l15) * GLT;

    for (int kt = 0; kt < K; kt += GKT) {
#ifdef USE_ASYNC_LDS
        __builtin_amdgcn_global_load_async_to_lds_b128(
            (AS1 v4i*)(ga + kt),  (AS3 v4i*)la,  0, 0);
        __builtin_amdgcn_global_load_async_to_lds_b128(
            (AS1 v4i*)(gb0 + kt), (AS3 v4i*)lb0, 0, 0);
        __builtin_amdgcn_global_load_async_to_lds_b128(
            (AS1 v4i*)(gb1 + kt), (AS3 v4i*)lb1, 0, 0);
        asm volatile("s_wait_asynccnt 0x0" ::: "memory");
#else
        *(uint4*)la  = *(const uint4*)(ga + kt);
        *(uint4*)lb0 = *(const uint4*)(gb0 + kt);
        *(uint4*)lb1 = *(const uint4*)(gb1 + kt);
#endif
        __syncthreads();

        FragBF fa;
        #pragma unroll
        for (int v = 0; v < 8; ++v) {
            // A 16x32 bf16 layout: v0..3 K=hi*8+2v, v4..7 K=16+hi*8+2(v-4)
            int ka = ((v & 4) ? 16 : 0) + (hi ? 8 : 0) + (v & 3) * 2;
            fa.u[v] = *(const unsigned*)&As[arow + ka];
        }
        #pragma unroll
        for (int g = 0; g < 4; ++g) {
            FragBF fb;
            #pragma unroll
            for (int v = 0; v < 8; ++v) {
                // B 32x16 bf16 layout: v0..3 K=hi*8+2v, v4..7 K=16+hi*8+2(v-4)
                int kbv = ((v & 4) ? 16 : 0) + (hi ? 8 : 0) + (v & 3) * 2;
                fb.u[v] = *(const unsigned*)&Bs[brow[g] + kbv];
            }
            acc[g] = __builtin_amdgcn_wmma_f32_16x16x32_bf16(
                         false, fa.v, false, fb.v, (short)0, acc[g], false, false);
        }
        __syncthreads();
    }

    // C/D layout: lane = column (n), vgpr r = row (+8 for upper half-wave)
    #pragma unroll
    for (int g = 0; g < 4; ++g) {
        const int gn = n0 + wn * 64 + g * 16 + l15;
        if (gn < N) {
            #pragma unroll
            for (int r = 0; r < 8; ++r) {
                int gm = m0 + wm * 16 + hi * 8 + r;
                float v0 = acc[g][r];
                if (mode == 1) v0 = softplusf(v0 + bias[gn]);
                C[(size_t)gm * ldc + gn] = v0;
                if (Cbf && gn < Nbf) Cbf[(size_t)gm * ldcbf + gn] = f2bf(v0);
            }
        }
    }
}

// ---------------------------------------------------------------------------
// Residual add + LayerNorm (D=1024), one token per block. Also emits bf16 copy
// of the normalized output (feeds in_proj GEMM) when out_bf != null.
// ---------------------------------------------------------------------------
__global__ __launch_bounds__(256)
void add_ln_kernel(const float* __restrict__ x, float* __restrict__ resid,
                   const float* __restrict__ w, const float* __restrict__ bb,
                   float* __restrict__ out, unsigned short* __restrict__ out_bf,
                   int addflag)
{
    __shared__ float s1[256];
    __shared__ float s2[256];
    const int t = blockIdx.x, tid = threadIdx.x;
    const int base = t * 1024 + tid * 4;

    float v[4];
    float s = 0.f, sq = 0.f;
    #pragma unroll
    for (int i = 0; i < 4; ++i) {
        float r = x[base + i];
        if (addflag) r += resid[base + i];
        resid[base + i] = r;
        v[i] = r; s += r; sq += r * r;
    }
    s1[tid] = s; s2[tid] = sq;
    __syncthreads();
    for (int off = 128; off > 0; off >>= 1) {
        if (tid < off) { s1[tid] += s1[tid + off]; s2[tid] += s2[tid + off]; }
        __syncthreads();
    }
    const float mu  = s1[0] * (1.f / 1024.f);
    const float var = s2[0] * (1.f / 1024.f) - mu * mu;
    const float rs  = rsqrtf(var + 1e-5f);
    float o[4];
    #pragma unroll
    for (int i = 0; i < 4; ++i) {
        int c = tid * 4 + i;
        o[i] = (v[i] - mu) * rs * w[c] + bb[c];
        out[base + i] = o[i];
    }
    if (out_bf) {
        *(unsigned*)&out_bf[base + 0] = pack_bf16(o[0], o[1]);
        *(unsigned*)&out_bf[base + 2] = pack_bf16(o[2], o[3]);
    }
}

// ---------------------------------------------------------------------------
// Depthwise causal conv (D_CONV=4) + SiLU. Emits f32 (for scan) + bf16 (GEMM A)
// ---------------------------------------------------------------------------
__global__ __launch_bounds__(256)
void conv_silu_kernel(const float* __restrict__ xz, const float* __restrict__ cw,
                      const float* __restrict__ cb, float* __restrict__ xc,
                      unsigned short* __restrict__ xc_bf)
{
    const int idx = blockIdx.x * 256 + threadIdx.x;
    const int d = idx & 2047;
    const int l = (idx >> 11) & 1023;
    const int b = idx >> 21;
    float acc = cb[d];
    #pragma unroll
    for (int k = 0; k < 4; ++k) {
        int lk = l + k - 3;
        if (lk >= 0) acc += xz[((size_t)(b * 1024 + lk)) * 4096 + d] * cw[d * 4 + k];
    }
    float r = acc / (1.f + __expf(-acc));   // SiLU
    xc[idx]    = r;
    xc_bf[idx] = f2bf(r);
}

// ---------------------------------------------------------------------------
// Selective scan: sequential over L, parallel over (b,d). 16 states/channel in
// registers; B_t/C_t staged in LDS per 64-step chunk (+prefetch of next chunk).
// Fuses +x*D and *silu(z); emits y directly as bf16 for out_proj.
// ---------------------------------------------------------------------------
__global__ __launch_bounds__(64)
void scan_kernel(const float* __restrict__ dtb, const float* __restrict__ xcb,
                 const float* __restrict__ xz,  const float* __restrict__ dbl,
                 const float* __restrict__ Alog, const float* __restrict__ Dv,
                 unsigned short* __restrict__ ybf)
{
    const int L = 1024;
    const int b = blockIdx.x >> 5;
    const int d = (blockIdx.x & 31) * 64 + threadIdx.x;

    float Ac[16];
    #pragma unroll
    for (int s = 0; s < 16; ++s) Ac[s] = -__expf(Alog[d * 16 + s]);
    const float Dp = Dv[d];

    float h[16];
    #pragma unroll
    for (int s = 0; s < 16; ++s) h[s] = 0.f;

    __shared__ float BC[64 * 32];   // 64 timesteps x (16 B + 16 C)

    for (int tc = 0; tc < L; tc += 64) {
        for (int e = threadIdx.x; e < 64 * 32; e += 64) {
            int tr = e >> 5, c = e & 31;
            BC[e] = dbl[(size_t)(b * L + tc + tr) * 96 + 64 + c];
            if (tc + 64 < L)  // prefetch next chunk's B/C rows into cache
                __builtin_prefetch(&dbl[(size_t)(b * L + tc + 64 + tr) * 96 + 64 + c], 0, 1);
        }
        __syncthreads();
        for (int ti = 0; ti < 64; ++ti) {
            const int t = tc + ti;
            const size_t idx = (size_t)(b * L + t) * 2048 + d;
            const float dtv = dtb[idx];
            const float xv  = xcb[idx];
            const float zv  = xz[(size_t)(b * L + t) * 4096 + 2048 + d];
            const float dtx = dtv * xv;
            const float* R  = &BC[ti * 32];
            float y = 0.f;
            #pragma unroll
            for (int s = 0; s < 16; ++s) {
                float dA = __expf(dtv * Ac[s]);
                h[s] = h[s] * dA + dtx * R[s];
                y += h[s] * R[16 + s];
            }
            y += xv * Dp;
            const float sz = zv / (1.f + __expf(-zv));
            ybf[idx] = f2bf(y * sz);
        }
        __syncthreads();
    }
}

// ---------------------------------------------------------------------------
// Host-side pipeline
// ---------------------------------------------------------------------------
extern "C" void kernel_launch(void* const* d_in, const int* in_sizes, int n_in,
                              void* d_out, int out_size, void* d_ws, size_t ws_size,
                              hipStream_t stream)
{
    (void)in_sizes; (void)n_in; (void)out_size; (void)ws_size;

    const float* hsin = (const float*)d_in[0];   // (2,1024,1024)
    const float* ipw  = (const float*)d_in[1];   // (4,4096,1024)
    const float* cw   = (const float*)d_in[2];   // (4,2048,4)
    const float* cb   = (const float*)d_in[3];   // (4,2048)
    const float* xpw  = (const float*)d_in[4];   // (4,96,2048)
    const float* dpw  = (const float*)d_in[5];   // (4,2048,64)
    const float* dpb  = (const float*)d_in[6];   // (4,2048)
    const float* Alog = (const float*)d_in[7];   // (4,2048,16)
    const float* Dv   = (const float*)d_in[8];   // (4,2048)
    const float* opw  = (const float*)d_in[9];   // (4,1024,2048)
    const float* nw   = (const float*)d_in[10];  // (4,1024)
    const float* nb   = (const float*)d_in[11];  // (4,1024)
    const float* nfw  = (const float*)d_in[12];  // (1024)
    const float* nfb  = (const float*)d_in[13];  // (1024)
    float* outp = (float*)d_out;

    const int M = 2048;  // B*L tokens

    // ---- workspace carve-up ----
    float* ws    = (float*)d_ws;
    float* resid = ws;                         // 2048*1024
    float* hs    = resid + 2048 * 1024;        // 2048*1024
    float* xzb   = hs    + 2048 * 1024;        // 2048*4096
    float* xcb   = xzb   + 2048 * 4096;        // 2048*2048
    float* dblb  = xcb   + 2048 * 2048;        // 2048*96
    float* dtb   = dblb  + 2048 * 96;          // 2048*2048
    unsigned short* bfw = (unsigned short*)(dtb + 2048 * 2048);
    unsigned short* hs_bf   = bfw;                         // 2048*1024
    unsigned short* xc_bf   = hs_bf   + 2048 * 1024;       // 2048*2048
    unsigned short* dtin_bf = xc_bf   + 2048 * 2048;       // 2048*64
    unsigned short* y_bf    = dtin_bf + 2048 * 64;         // 2048*2048
    unsigned short* w_ip    = y_bf    + 2048 * 2048;       // 4*4096*1024
    unsigned short* w_xp    = w_ip    + 4 * 4096 * 1024;   // 4*96*2048
    unsigned short* w_dp    = w_xp    + 4 * 96 * 2048;     // 4*2048*64
    unsigned short* w_op    = w_dp    + 4 * 2048 * 64;     // 4*1024*2048

    // ---- weights: f32 -> bf16 once per launch (L2-resident afterwards) ----
    {
        int n2;
        n2 = 4 * 4096 * 1024 / 2;
        cvt_f32_bf16_kernel<<<(n2 + 255) / 256, 256, 0, stream>>>(ipw, w_ip, n2);
        n2 = 4 * 96 * 2048 / 2;
        cvt_f32_bf16_kernel<<<(n2 + 255) / 256, 256, 0, stream>>>(xpw, w_xp, n2);
        n2 = 4 * 2048 * 64 / 2;
        cvt_f32_bf16_kernel<<<(n2 + 255) / 256, 256, 0, stream>>>(dpw, w_dp, n2);
        n2 = 4 * 1024 * 2048 / 2;
        cvt_f32_bf16_kernel<<<(n2 + 255) / 256, 256, 0, stream>>>(opw, w_op, n2);
    }

    for (int i = 0; i < 4; ++i) {
        // residual + LayerNorm (emits f32 + bf16)
        add_ln_kernel<<<2048, 256, 0, stream>>>(
            (i == 0) ? hsin : hs, resid, nw + i * 1024, nb + i * 1024,
            hs, hs_bf, (i > 0) ? 1 : 0);

        // in_proj: (2048x1024) x (4096x1024)^T -> xz f32
        {
            dim3 g(M / 64, 4096 / 128);
            gemm_bf16_wmma<<<g, 256, 0, stream>>>(hs_bf, 1024,
                w_ip + (size_t)i * 4096 * 1024, nullptr,
                xzb, 4096, nullptr, 0, 0, M, 4096, 1024, 0);
        }

        // depthwise conv + SiLU (emits f32 + bf16)
        conv_silu_kernel<<<(2 * 1024 * 2048) / 256, 256, 0, stream>>>(
            xzb, cw + (size_t)i * 2048 * 4, cb + i * 2048, xcb, xc_bf);

        // x_proj: (2048x2048) x (96x2048)^T -> dbl f32 ; dt slice also as bf16
        {
            dim3 g(M / 64, (96 + 127) / 128);
            gemm_bf16_wmma<<<g, 256, 0, stream>>>(xc_bf, 2048,
                w_xp + (size_t)i * 96 * 2048, nullptr,
                dblb, 96, dtin_bf, 64, 64, M, 96, 2048, 0);
        }

        // dt_proj + softplus: (2048x64) x (2048x64)^T -> dt f32
        {
            dim3 g(M / 64, 2048 / 128);
            gemm_bf16_wmma<<<g, 256, 0, stream>>>(dtin_bf, 64,
                w_dp + (size_t)i * 2048 * 64, dpb + i * 2048,
                dtb, 2048, nullptr, 0, 0, M, 2048, 64, 1);
        }

        // selective scan (fused +x*D, *silu(z)); emits y as bf16
        scan_kernel<<<64, 64, 0, stream>>>(dtb, xcb, xzb, dblb,
                                           Alog + (size_t)i * 2048 * 16,
                                           Dv + i * 2048, y_bf);

        // out_proj: (2048x2048) x (1024x2048)^T -> hs f32
        {
            dim3 g(M / 64, 1024 / 128);
            gemm_bf16_wmma<<<g, 256, 0, stream>>>(y_bf, 2048,
                w_op + (size_t)i * 1024 * 2048, nullptr,
                hs, 1024, nullptr, 0, 0, M, 1024, 2048, 0);
        }
    }

    // final residual + LayerNorm -> output
    add_ln_kernel<<<2048, 256, 0, stream>>>(hs, resid, nfw, nfb, outp, nullptr, 1);
}
